// SpikingRWKVBlock_33071248179957
// MI455X (gfx1250) — matrix operand order repaired and checked by reading, not compiled, not run on hardware
//
#include <hip/hip_runtime.h>
#include <hip/hip_bf16.h>

typedef __attribute__((ext_vector_type(16))) _Float16 v16h;
typedef __attribute__((ext_vector_type(8)))  _Float16 v8h;
typedef __attribute__((ext_vector_type(8)))  float    v8f;

// ---------------------------------------------------------------------------
// f32 -> f16 convert (streaming)
// ---------------------------------------------------------------------------
__global__ __launch_bounds__(256) void cvt_f32_f16(const float* __restrict__ in,
                                                   _Float16* __restrict__ out,
                                                   size_t n) {
    size_t i = (size_t)blockIdx.x * blockDim.x + threadIdx.x;
    if (i < n) out[i] = (_Float16)in[i];
}

// ---------------------------------------------------------------------------
// GEMM: C[M,N] = A16[M,K] * W16[N,K]^T (+ bias[N])   -- WMMA f16 -> f32 accum
// Block = 256 threads (8 waves), 128x128 C tile; wave = 64x32 (4x2 wmma tiles)
// A/B tiles (128x32 f16) staged in LDS via async-to-LDS, double buffered.
// LDS rows padded to 80B (40 halfs) so fragment reads hit distinct banks.
// ---------------------------------------------------------------------------
#define TILE_HALFS (128 * 40)          // one 128x32 tile, padded (10KB)
__global__ __launch_bounds__(256) void gemm_wmma_f16(
    const _Float16* __restrict__ A, const _Float16* __restrict__ W,
    const float* __restrict__ bias, float* __restrict__ C,
    int M, int N, int K)
{
    __shared__ _Float16 smem[4 * TILE_HALFS];   // [buf][A|B] = 40KB

    const int tid  = threadIdx.x;
    const int lane = tid & 31;
    const int w    = tid >> 5;
    const int wm   = w >> 2;        // 0..1 -> 64 rows each
    const int wn   = w & 3;         // 0..3 -> 32 cols each
    const int bM   = blockIdx.y * 128;
    const int bN   = blockIdx.x * 128;
    const int l16  = lane & 15;
    const int half = lane >> 4;

    const unsigned sbase = (unsigned)(unsigned long long)(size_t)&smem[0];

    // async-copy assignment: 512 16B transfers per tile, 2 per thread.
    // transfer t: row = t>>2, seg = t&3 (16B segment within 64B row)
    const int r0 = tid >> 2,          s0 = tid & 3;
    const int r1 = (tid + 256) >> 2,  s1 = (tid + 256) & 3;

    // per-thread global source pointers (advance by 32 halfs per k-step)
    const _Float16* gA0 = A + (size_t)(bM + r0) * K + s0 * 8;
    const _Float16* gA1 = A + (size_t)(bM + r1) * K + s1 * 8;
    const _Float16* gB0 = W + (size_t)(bN + r0) * K + s0 * 8;
    const _Float16* gB1 = W + (size_t)(bN + r1) * K + s1 * 8;
    // per-thread LDS destination offsets (within one buffer)
    const unsigned lA0 = (unsigned)(r0 * 80 + s0 * 16);
    const unsigned lA1 = (unsigned)(r1 * 80 + s1 * 16);
    const unsigned lB0 = lA0 + 2 * TILE_HALFS;   // bytes: TILE_HALFS halfs
    const unsigned lB1 = lA1 + 2 * TILE_HALFS;

    auto issue_tile = [&](int buf, int k0) {
        const unsigned bo = sbase + (unsigned)(buf * 4 * TILE_HALFS); // bytes/buffer = 2 tiles
        const _Float16* pa0 = gA0 + k0;
        const _Float16* pa1 = gA1 + k0;
        const _Float16* pb0 = gB0 + k0;
        const _Float16* pb1 = gB1 + k0;
        unsigned da0 = bo + lA0, da1 = bo + lA1, db0 = bo + lB0, db1 = bo + lB1;
        asm volatile("global_load_async_to_lds_b128 %0, %1, off" :: "v"(da0), "v"(pa0) : "memory");
        asm volatile("global_load_async_to_lds_b128 %0, %1, off" :: "v"(da1), "v"(pa1) : "memory");
        asm volatile("global_load_async_to_lds_b128 %0, %1, off" :: "v"(db0), "v"(pb0) : "memory");
        asm volatile("global_load_async_to_lds_b128 %0, %1, off" :: "v"(db1), "v"(pb1) : "memory");
    };

    v8f acc[4][2] = {};

    const int nIter = K >> 5;          // K/32
    issue_tile(0, 0);

    for (int it = 0; it < nIter; ++it) {
        const int cur = it & 1;
        if (it + 1 < nIter) {
            issue_tile(cur ^ 1, (it + 1) << 5);
            // 4 ops just issued for next buffer; <=4 outstanding ==> current
            // buffer's 4 ops (issued earlier, in-order completion) are done.
            asm volatile("s_wait_asynccnt 0x4" ::: "memory");
        } else {
            asm volatile("s_wait_asynccnt 0x0" ::: "memory");
        }
        __syncthreads();   // all waves' copies for `cur` complete

        const _Float16* sA = smem + cur * 2 * TILE_HALFS;
        const _Float16* sB = sA + TILE_HALFS;

        v16h a[4], b[2];
#pragma unroll
        for (int mt = 0; mt < 4; ++mt) {
            const int row = wm * 64 + mt * 16 + l16;
            const _Float16* p = sA + row * 40 + 8 * half;
            union { v16h v; v8h h[2]; } u;
            u.h[0] = *(const v8h*)p;          // A[m, 8h .. 8h+7]
            u.h[1] = *(const v8h*)(p + 16);   // A[m, 16+8h .. +7]
            a[mt] = u.v;
        }
#pragma unroll
        for (int nt = 0; nt < 2; ++nt) {
            const int row = wn * 32 + nt * 16 + l16;
            const _Float16* p = sB + row * 40 + 8 * half;
            union { v16h v; v8h h[2]; } u;
            u.h[0] = *(const v8h*)p;          // W[n, 8h .. 8h+7]
            u.h[1] = *(const v8h*)(p + 16);   // W[n, 16+8h .. +7]
            b[nt] = u.v;
        }
#pragma unroll
        for (int mt = 0; mt < 4; ++mt)
#pragma unroll
            for (int nt = 0; nt < 2; ++nt)
                acc[mt][nt] = __builtin_amdgcn_wmma_f32_16x16x32_f16(
                    false, a[mt], false, b[nt], (short)0, acc[mt][nt],
                    false, false);

        __syncthreads();   // done reading `cur` before it is overwritten
    }

    const int ncol0 = bN + wn * 32 + l16;
#pragma unroll
    for (int mt = 0; mt < 4; ++mt) {
#pragma unroll
        for (int nt = 0; nt < 2; ++nt) {
            const int col = ncol0 + nt * 16;
            const float bv = bias ? bias[col] : 0.0f;
#pragma unroll
            for (int r = 0; r < 8; ++r) {
                const int row = bM + wm * 64 + mt * 16 + r + 8 * half;
                C[(size_t)row * N + col] = acc[mt][nt][r] + bv;
            }
        }
    }
}

// ---------------------------------------------------------------------------
// RWKV spiking recurrence over T: one thread per (b,d) channel
// ---------------------------------------------------------------------------
__global__ __launch_bounds__(256) void rwkv_scan(
    const float* __restrict__ R, const float* __restrict__ Kk,
    const float* __restrict__ V, _Float16* __restrict__ Y16,
    int Bn, int T, int D)
{
    const int idx = blockIdx.x * blockDim.x + threadIdx.x;
    if (idx >= Bn * D) return;
    const int b = idx / D, d = idx % D;
    size_t base = (size_t)b * T * D + d;
    float vr = 0.f, vk = 0.f, vv = 0.f, h = 0.f;
    for (int t = 0; t < T; ++t) {
        const float r = R[base], k = Kk[base], v = V[base];
        vr += (r - vr) * 0.5f; const float sr = (vr >= 1.0f) ? 1.f : 0.f; vr *= (1.f - sr);
        vk += (k - vk) * 0.5f; const float sk = (vk >= 1.0f) ? 1.f : 0.f; vk *= (1.f - sk);
        vv += (v - vv) * 0.5f; const float sv = (vv >= 1.0f) ? 1.f : 0.f; vv *= (1.f - sv);
        h = h * 0.9f + sk * sv;
        Y16[base] = (_Float16)(sr * h);
        base += D;
    }
}

// ---------------------------------------------------------------------------
// LIF scan over T: one thread per (b,c) channel.  OutT = _Float16 or float.
// ---------------------------------------------------------------------------
template <typename OutT>
__global__ __launch_bounds__(256) void lif_scan_k(
    const float* __restrict__ X, OutT* __restrict__ S, int Bn, int T, int C)
{
    const int idx = blockIdx.x * blockDim.x + threadIdx.x;
    if (idx >= Bn * C) return;
    const int b = idx / C, c = idx % C;
    size_t base = (size_t)b * T * C + c;
    float v = 0.f;
    for (int t = 0; t < T; ++t) {
        const float x = X[base];
        v += (x - v) * 0.5f;
        const float s = (v >= 1.0f) ? 1.f : 0.f;
        v *= (1.f - s);
        S[base] = (OutT)s;
        base += C;
    }
}

// ---------------------------------------------------------------------------
// LayerNorm of (a + b) over last dim; writes f32 and optionally f16 copy.
// One block (256 thr) per row, D multiple of 256, D/256 <= 8.
// ---------------------------------------------------------------------------
__global__ __launch_bounds__(256) void ln_kernel(
    const float* __restrict__ A, const float* __restrict__ Bb,
    const float* __restrict__ g, const float* __restrict__ beta,
    float* __restrict__ out32, _Float16* __restrict__ out16, int D)
{
    __shared__ float ssum[256];
    __shared__ float ssq[256];
    const int row = blockIdx.x;
    const int tid = threadIdx.x;
    const size_t base = (size_t)row * D;
    const int nPer = D >> 8;   // D/256

    float vals[8];
    float s = 0.f, q = 0.f;
    for (int i = 0; i < nPer; ++i) {
        const int c = tid + (i << 8);
        const float v = A[base + c] + Bb[base + c];
        vals[i] = v; s += v; q += v * v;
    }
    ssum[tid] = s; ssq[tid] = q;
    __syncthreads();
    for (int off = 128; off > 0; off >>= 1) {
        if (tid < off) { ssum[tid] += ssum[tid + off]; ssq[tid] += ssq[tid + off]; }
        __syncthreads();
    }
    const float mu   = ssum[0] / D;
    float var        = ssq[0] / D - mu * mu;
    if (var < 0.f) var = 0.f;
    const float rstd = rsqrtf(var + 1e-5f);
    for (int i = 0; i < nPer; ++i) {
        const int c = tid + (i << 8);
        const float o = (vals[i] - mu) * rstd * g[c] + beta[c];
        out32[base + c] = o;
        if (out16) out16[base + c] = (_Float16)o;
    }
}

// ---------------------------------------------------------------------------
// Launch
// ---------------------------------------------------------------------------
extern "C" void kernel_launch(void* const* d_in, const int* in_sizes, int n_in,
                              void* d_out, int out_size, void* d_ws, size_t ws_size,
                              hipStream_t stream) {
    const float* x    = (const float*)d_in[0];
    const float* Wr   = (const float*)d_in[1];
    const float* Wk   = (const float*)d_in[2];
    const float* Wv   = (const float*)d_in[3];
    const float* Wo   = (const float*)d_in[4];
    const float* W1   = (const float*)d_in[5];
    const float* b1   = (const float*)d_in[6];
    const float* W2   = (const float*)d_in[7];
    const float* b2   = (const float*)d_in[8];
    const float* ln1g = (const float*)d_in[9];
    const float* ln1b = (const float*)d_in[10];
    const float* ln2g = (const float*)d_in[11];
    const float* ln2b = (const float*)d_in[12];

    const int D = in_sizes[9];            // 1024
    const int F = in_sizes[6];            // 4096
    const int M = in_sizes[0] / D;        // B*T = 8192
    const int Bn = 8;                     // batch (fixed by harness)
    const int T = M / Bn;                 // 1024

    // ---- workspace arena ----
    char* p = (char*)d_ws;
    const size_t szDD = (size_t)D * D * sizeof(_Float16);
    const size_t szDF = (size_t)D * F * sizeof(_Float16);
    _Float16* wr16 = (_Float16*)p; p += szDD;
    _Float16* wk16 = (_Float16*)p; p += szDD;
    _Float16* wv16 = (_Float16*)p; p += szDD;
    _Float16* wo16 = (_Float16*)p; p += szDD;
    _Float16* w116 = (_Float16*)p; p += szDF;
    _Float16* w216 = (_Float16*)p; p += szDF;
    _Float16* act16 = (_Float16*)p; p += (size_t)M * F * sizeof(_Float16); // 64MB, recycled
    float* bufP = (float*)p;        p += (size_t)M * F * sizeof(float);    // 128MB: R|K|V then h1
    float* bufQ = (float*)p;        p += (size_t)M * D * sizeof(float);    // 32MB: attn then h2/ffn
    float* x1   = (float*)p;        p += (size_t)M * D * sizeof(float);    // 32MB
    (void)ws_size; (void)n_in;

    float* Rb = bufP;
    float* Kb = bufP + (size_t)M * D;
    float* Vb = bufP + (size_t)2 * M * D;

    dim3 blk(256);
    auto cvtGrid = [](size_t n) { return dim3((unsigned)((n + 255) / 256)); };

    // 0) converts: x and the six weights to f16
    cvt_f32_f16<<<cvtGrid((size_t)M * D), blk, 0, stream>>>(x,  act16, (size_t)M * D);
    cvt_f32_f16<<<cvtGrid((size_t)D * D), blk, 0, stream>>>(Wr, wr16, (size_t)D * D);
    cvt_f32_f16<<<cvtGrid((size_t)D * D), blk, 0, stream>>>(Wk, wk16, (size_t)D * D);
    cvt_f32_f16<<<cvtGrid((size_t)D * D), blk, 0, stream>>>(Wv, wv16, (size_t)D * D);
    cvt_f32_f16<<<cvtGrid((size_t)D * D), blk, 0, stream>>>(Wo, wo16, (size_t)D * D);
    cvt_f32_f16<<<cvtGrid((size_t)F * D), blk, 0, stream>>>(W1, w116, (size_t)F * D);
    cvt_f32_f16<<<cvtGrid((size_t)D * F), blk, 0, stream>>>(W2, w216, (size_t)D * F);

    // 1) R,K,V projections: [M,D] x [D,D]^T
    {
        dim3 g(D / 128, M / 128);
        gemm_wmma_f16<<<g, blk, 0, stream>>>(act16, wr16, nullptr, Rb, M, D, D);
        gemm_wmma_f16<<<g, blk, 0, stream>>>(act16, wk16, nullptr, Kb, M, D, D);
        gemm_wmma_f16<<<g, blk, 0, stream>>>(act16, wv16, nullptr, Vb, M, D, D);
    }

    // 2) RWKV spiking scan -> y (f16, overwrites act16; x16 no longer needed)
    rwkv_scan<<<dim3((Bn * D + 255) / 256), blk, 0, stream>>>(Rb, Kb, Vb, act16, Bn, T, D);

    // 3) attn = y @ Wo^T
    {
        dim3 g(D / 128, M / 128);
        gemm_wmma_f16<<<g, blk, 0, stream>>>(act16, wo16, nullptr, bufQ, M, D, D);
    }

    // 4) x1 = LN(x + attn); also emit f16 copy for the fc1 GEMM
    ln_kernel<<<dim3(M), blk, 0, stream>>>(x, bufQ, ln1g, ln1b, x1, act16, D);

    // 5) h1 = x1 @ W1^T + b1   [M,F]  (bufP recycled: R/K/V dead)
    {
        dim3 g(F / 128, M / 128);
        gemm_wmma_f16<<<g, blk, 0, stream>>>(act16, w116, b1, bufP, M, F, D);
    }

    // 6) s1 = LIF scan over h1 -> f16 spikes (act16 recycled)
    lif_scan_k<_Float16><<<dim3((Bn * F + 255) / 256), blk, 0, stream>>>(bufP, act16, Bn, T, F);

    // 7) h2 = s1 @ W2^T + b2   [M,D]  (bufQ recycled: attn dead)
    {
        dim3 g(D / 128, M / 128);
        gemm_wmma_f16<<<g, blk, 0, stream>>>(act16, w216, b2, bufQ, M, D, F);
    }

    // 8) ffn = LIF scan over h2 (in place: each element read once before write)
    lif_scan_k<float><<<dim3((Bn * D + 255) / 256), blk, 0, stream>>>(bufQ, bufQ, Bn, T, D);

    // 9) out = LN(x1 + ffn)
    ln_kernel<<<dim3(M), blk, 0, stream>>>(x1, bufQ, ln2g, ln2b, (float*)d_out, nullptr, D);
}
